// Head_3813930959439
// MI455X (gfx1250) — matrix operand order repaired and checked by reading, not compiled, not run on hardware
//
#include <hip/hip_runtime.h>
#include <hip/hip_bf16.h>

typedef __attribute__((ext_vector_type(16))) __bf16 v16bf;
typedef __attribute__((ext_vector_type(8)))  float  v8f;

union BF16x16 { v16bf v; unsigned int u32[8]; unsigned short u16[16]; };

#define HH    48
#define WW    96
#define HWSZ  4608          // 48*96
#define HEADS 4
#define DH    128
#define QSCALE 0.08838834764831845f   // 128^-0.5
#define EPSV  1e-5f

#if __has_builtin(__builtin_amdgcn_sched_barrier)
#define SCHED_FENCE() __builtin_amdgcn_sched_barrier(0)
#else
#define SCHED_FENCE() asm volatile("" ::: "memory")
#endif

__device__ __forceinline__ unsigned short f2bf(float f) {
  unsigned int u = __float_as_uint(f);
  u += 0x7FFFu + ((u >> 16) & 1u);          // round-to-nearest-even
  return (unsigned short)(u >> 16);
}
__device__ __forceinline__ float bf2f(unsigned short b) {
  return __uint_as_float(((unsigned int)b) << 16);
}

// ---------------------------------------------------------------------------
// Kernel 1: h1 = relu(bn1(w_in @ x)), stored [64][4608]
// grid (4608/256, 64)
__global__ void k_proj_in(const float* __restrict__ x, const float* __restrict__ w_in,
                          const float* __restrict__ g, const float* __restrict__ b,
                          const float* __restrict__ m, const float* __restrict__ v,
                          float* __restrict__ h1) {
  const int i = blockIdx.x * blockDim.x + threadIdx.x;   // pixel
  const int c = blockIdx.y;                              // out channel (0..63)
  float acc = 0.f;
  #pragma unroll 4
  for (int ic = 0; ic < 256; ++ic)
    acc = fmaf(x[(size_t)ic * HWSZ + i], w_in[c * 256 + ic], acc);
  const float inv = g[c] * rsqrtf(v[c] + EPSV);
  float r = (acc - m[c]) * inv + b[c];
  h1[(size_t)c * HWSZ + i] = r > 0.f ? r : 0.f;
}

// ---------------------------------------------------------------------------
// Kernel 2: qkv = w_qkv @ h1 -> q(scaled)/k/v in bf16, layout [head][i][128]
// grid (4608/256, 1536)
__global__ void k_qkv(const float* __restrict__ h1, const float* __restrict__ w_qkv,
                      unsigned short* __restrict__ qb, unsigned short* __restrict__ kb,
                      unsigned short* __restrict__ vb) {
  const int i = blockIdx.x * blockDim.x + threadIdx.x;
  const int o = blockIdx.y;                              // 0..1535
  float acc = 0.f;
  #pragma unroll 8
  for (int c = 0; c < 64; ++c)
    acc = fmaf(h1[(size_t)c * HWSZ + i], w_qkv[o * 64 + c], acc);
  const int s = o >> 9;            // 0:q 1:k 2:v
  const int h = (o >> 7) & 3;
  const int d = o & 127;
  unsigned short* dst = (s == 0) ? qb : (s == 1 ? kb : vb);
  if (s == 0) acc *= QSCALE;
  dst[((size_t)h * HWSZ + i) * DH + d] = f2bf(acc);
}

// ---------------------------------------------------------------------------
// Kernel 3: factorized relative position bias tables.
// RW[h][i][wk] = q_i . rel_w[wk - w_i + 95],  RH[h][i][hk] = q_i . rel_h[hk - h_i + 47]
// grid (4608/256, 4*144): y = h*144 + r, r<96 -> RW col, r>=96 -> RH col
__global__ void k_relbias(const unsigned short* __restrict__ qb,
                          const float* __restrict__ rel_h, const float* __restrict__ rel_w,
                          float* __restrict__ RW, float* __restrict__ RH) {
  const int i = blockIdx.x * blockDim.x + threadIdx.x;
  const int h = blockIdx.y / 144;
  const int r = blockIdx.y % 144;
  const unsigned short* q = qb + ((size_t)h * HWSZ + i) * DH;
  float acc = 0.f;
  if (r < 96) {
    const int wi = i % WW;
    const float* rk = rel_w + (size_t)(r - wi + WW - 1) * DH;
    #pragma unroll 8
    for (int d = 0; d < DH; ++d) acc = fmaf(bf2f(q[d]), rk[d], acc);
    RW[((size_t)h * HWSZ + i) * WW + r] = acc;
  } else {
    const int hk = r - 96;
    const int hi = i / WW;
    const float* rk = rel_h + (size_t)(hk - hi + HH - 1) * DH;
    #pragma unroll 8
    for (int d = 0; d < DH; ++d) acc = fmaf(bf2f(q[d]), rk[d], acc);
    RH[((size_t)h * HWSZ + i) * HH + hk] = acc;
  }
}

// ---------------------------------------------------------------------------
// Kernel 4: flash attention. 288 blocks (head, 64-query chunk) x 128 threads.
// Each wave32 owns a 16-query tile; loops over 144 key tiles of 32.
// K tile is async-copied global->LDS (ASYNCcnt path). B fragments are batch-
// preloaded behind a sched_barrier so the v_wmma ops issue back-to-back.
__global__ __launch_bounds__(128)
void k_attn(const unsigned short* __restrict__ qb, const unsigned short* __restrict__ kb,
            const unsigned short* __restrict__ vb, const float* __restrict__ RW,
            const float* __restrict__ RH, float* __restrict__ attn_t) {
  __shared__ unsigned short Kt[32][136];    // [j][d], padded row -> no bank conflicts
  __shared__ unsigned short Vt[128][40];    // transposed [d][j], padded
  __shared__ unsigned short Pws[4][16][40]; // per-wave P re-lane staging

  const int blk  = blockIdx.x;              // 288
  const int h    = blk / 72;
  const int q0b  = (blk % 72) * 64;
  const int tid  = threadIdx.x;
  const int wave = tid >> 5;
  const int lane = tid & 31;
  const int lm   = lane & 15;               // 0..15
  const int lg   = lane >> 4;               // 0/1 lane-half
  const int qrow0 = q0b + wave * 16;

  const unsigned short* qbase = qb + (size_t)h * HWSZ * DH;
  const unsigned short* kbase = kb + (size_t)h * HWSZ * DH;
  const unsigned short* vbase = vb + (size_t)h * HWSZ * DH;
  const float* RWb = RW + (size_t)h * HWSZ * WW;
  const float* RHb = RH + (size_t)h * HWSZ * HH;

  // Q tile (16x128) resident in registers, ISA A-layout, 4 chunks of K=32.
  BF16x16 qf[4];
  #pragma unroll
  for (int c = 0; c < 4; ++c) {
    #pragma unroll
    for (int t = 0; t < 8; ++t) {
      const int kk = ((t < 4) ? 0 : 16) + 8 * lg + 2 * (t & 3);
      qf[c].u32[t] = *(const unsigned int*)(qbase + ((size_t)(qrow0 + lm)) * DH + 32 * c + kk);
    }
  }

  v8f O[8];
  float mrow[8], lrow[8];
  #pragma unroll
  for (int dg = 0; dg < 8; ++dg) {
    #pragma unroll
    for (int r = 0; r < 8; ++r) O[dg][r] = 0.f;
  }
  #pragma unroll
  for (int r = 0; r < 8; ++r) { mrow[r] = -1e30f; lrow[r] = 0.f; }

  for (int jt = 0; jt < 144; ++jt) {
    const int j0 = jt * 32;
    __syncthreads();   // all waves done with previous K/V tile

    // ---- K tile: async global->LDS copy (no VGPR staging, ASYNCcnt-tracked)
    // 512 x 16B chunks, 4 per thread. Row stride in LDS is 136 shorts.
    #pragma unroll
    for (int s = 0; s < 4; ++s) {
      const int c   = s * 128 + tid;            // chunk id
      const int j   = c >> 4;                   // row (16 chunks per 256B row)
      const int c16 = c & 15;
      const unsigned lds = (unsigned)(size_t)&Kt[j][c16 * 8];
      const unsigned long long ga =
          (unsigned long long)(const void*)(kbase + (size_t)(j0 + j) * DH + (size_t)c16 * 8);
      asm volatile("global_load_async_to_lds_b128 %0, %1, off"
                   :: "v"(lds), "v"(ga) : "memory");
    }

    // ---- V tile: needs an element transpose, so go through VGPRs into Vt
    #pragma unroll
    for (int s = 0; s < 16; ++s) {
      const int idx = s * 128 + tid;       // 2048 u32
      const int j = idx >> 6, du = idx & 63;
      const unsigned int w =
          *(const unsigned int*)(vbase + (size_t)(j0 + j) * DH + du * 2);
      Vt[du * 2][j]     = (unsigned short)(w & 0xFFFFu);
      Vt[du * 2 + 1][j] = (unsigned short)(w >> 16);
    }

    // prefetch next K/V tile (64 x 128B lines each)
    if (jt + 1 < 144) {
      const unsigned short* src = (tid < 64) ? kbase : vbase;
      __builtin_prefetch(src + (size_t)(j0 + 32) * DH + (size_t)(tid & 63) * 64, 0, 1);
    }

    asm volatile("s_wait_asynccnt 0x0" ::: "memory");  // own async K chunks done
    __syncthreads();                                   // tiles visible to all waves

    // ---- S(16x32) = Q K^T: batch-preload all 8 B fragments, fence, then
    //      8 WMMAs back-to-back (single dscnt wait instead of one per WMMA).
    BF16x16 Bk0[4], Bk1[4];
    #pragma unroll
    for (int c = 0; c < 4; ++c) {
      #pragma unroll
      for (int t = 0; t < 8; ++t) {
        const int kk = ((t < 4) ? 0 : 16) + 8 * lg + 2 * (t & 3);  // B layout: k = (r<4?0:16) + 8g + 2(r&3) + h
        Bk0[c].u32[t] = *(const unsigned int*)&Kt[lm][32 * c + kk];
        Bk1[c].u32[t] = *(const unsigned int*)&Kt[16 + lm][32 * c + kk];
      }
    }
    SCHED_FENCE();   // keep the ds_load batch above the WMMA chain
    v8f S0 = {0.f,0.f,0.f,0.f,0.f,0.f,0.f,0.f};
    v8f S1 = {0.f,0.f,0.f,0.f,0.f,0.f,0.f,0.f};
    #pragma unroll
    for (int c = 0; c < 4; ++c) {
      S0 = __builtin_amdgcn_wmma_f32_16x16x32_bf16(false, qf[c].v, false, Bk0[c].v, (short)0, S0, false, false);
      S1 = __builtin_amdgcn_wmma_f32_16x16x32_bf16(false, qf[c].v, false, Bk1[c].v, (short)0, S1, false, false);
    }

    // factorized relative position bias: S[i][j] += RW[i][w_j] + RH[i][h_j]
    const int jA = j0 + lm,        jB = j0 + 16 + lm;
    const int wA = jA % WW, hA = jA / WW, wB = jB % WW, hB = jB / WW;
    #pragma unroll
    for (int r = 0; r < 8; ++r) {
      const int qi = qrow0 + r + 8 * lg;             // C/D layout: row = r + 8g
      S0[r] += RWb[(size_t)qi * WW + wA] + RHb[(size_t)qi * HH + hA];
      S1[r] += RWb[(size_t)qi * WW + wB] + RHb[(size_t)qi * HH + hB];
    }

    // online softmax (row reductions across the 16-lane half owning each row)
    #pragma unroll
    for (int r = 0; r < 8; ++r) {
      float mx = fmaxf(S0[r], S1[r]);
      #pragma unroll
      for (int off = 8; off; off >>= 1) mx = fmaxf(mx, __shfl_xor(mx, off, 16));
      const float mnew = fmaxf(mrow[r], mx);
      const float corr = __expf(mrow[r] - mnew);
      const float p0 = __expf(S0[r] - mnew);
      const float p1 = __expf(S1[r] - mnew);
      float ls = p0 + p1;
      #pragma unroll
      for (int off = 8; off; off >>= 1) ls += __shfl_xor(ls, off, 16);
      lrow[r] = lrow[r] * corr + ls;
      mrow[r] = mnew;
      S0[r] = p0; S1[r] = p1;
      #pragma unroll
      for (int dg = 0; dg < 8; ++dg) O[dg][r] *= corr;
    }

    // re-lane P (C/D layout -> A layout) through wave-private LDS
    #pragma unroll
    for (int r = 0; r < 8; ++r) {
      const int mm = r + 8 * lg;
      Pws[wave][mm][lm]      = f2bf(S0[r]);
      Pws[wave][mm][16 + lm] = f2bf(S1[r]);
    }
    asm volatile("s_wait_dscnt 0x0" ::: "memory");     // wave-internal LDS RAW
    BF16x16 Pf;
    #pragma unroll
    for (int t = 0; t < 8; ++t) {
      const int kk = ((t < 4) ? 0 : 16) + 8 * lg + 2 * (t & 3);
      Pf.u32[t] = *(const unsigned int*)&Pws[wave][lm][kk];
    }

    // ---- O(16x128) += P(16x32) V(32x128): batch-preload 8 B fragments,
    //      fence, 8 WMMAs back-to-back.
    BF16x16 Bv[8];
    #pragma unroll
    for (int dg = 0; dg < 8; ++dg) {
      #pragma unroll
      for (int t = 0; t < 8; ++t) {
        const int jj = ((t < 4) ? 0 : 16) + 8 * lg + 2 * (t & 3);
        Bv[dg].u32[t] = *(const unsigned int*)&Vt[16 * dg + lm][jj];
      }
    }
    SCHED_FENCE();
    #pragma unroll
    for (int dg = 0; dg < 8; ++dg)
      O[dg] = __builtin_amdgcn_wmma_f32_16x16x32_bf16(false, Pf.v, false, Bv[dg].v, (short)0, O[dg], false, false);
  }

  // finalize and store in [channel][pixel] layout (channel = h*128 + d)
  #pragma unroll
  for (int dg = 0; dg < 8; ++dg) {
    #pragma unroll
    for (int r = 0; r < 8; ++r) {
      const int mm = r + 8 * lg;
      const int ch = h * DH + 16 * dg + lm;
      attn_t[(size_t)ch * HWSZ + (qrow0 + mm)] = O[dg][r] / lrow[r];
    }
  }
}

// ---------------------------------------------------------------------------
// Kernel 5: fm = relu(bn2(w_out @ attn) + x), stored [256][4608]
// grid (4608/256, 256)
__global__ void k_out_proj(const float* __restrict__ attn_t, const float* __restrict__ w_out,
                           const float* __restrict__ g, const float* __restrict__ b,
                           const float* __restrict__ m, const float* __restrict__ v,
                           const float* __restrict__ x, float* __restrict__ fm) {
  const int i = blockIdx.x * blockDim.x + threadIdx.x;
  const int c = blockIdx.y;
  float acc = 0.f;
  #pragma unroll 4
  for (int k = 0; k < 512; ++k)
    acc = fmaf(attn_t[(size_t)k * HWSZ + i], w_out[c * 512 + k], acc);
  const float inv = g[c] * rsqrtf(v[c] + EPSV);
  float r = (acc - m[c]) * inv + b[c] + x[(size_t)c * HWSZ + i];
  fm[(size_t)c * HWSZ + i] = r > 0.f ? r : 0.f;
}

// ---------------------------------------------------------------------------
// Kernel 6: out = w_head @ fm + b_head, layout (19, 48, 96)
// grid (4608/256, 19)
__global__ void k_head(const float* __restrict__ fm, const float* __restrict__ w_head,
                       const float* __restrict__ b_head, float* __restrict__ out) {
  const int i  = blockIdx.x * blockDim.x + threadIdx.x;
  const int oc = blockIdx.y;
  float acc = b_head[oc];
  #pragma unroll 4
  for (int c = 0; c < 256; ++c)
    acc = fmaf(fm[(size_t)c * HWSZ + i], w_head[oc * 256 + c], acc);
  out[(size_t)oc * HWSZ + i] = acc;
}

// ---------------------------------------------------------------------------
extern "C" void kernel_launch(void* const* d_in, const int* in_sizes, int n_in,
                              void* d_out, int out_size, void* d_ws, size_t ws_size,
                              hipStream_t stream) {
  (void)in_sizes; (void)n_in; (void)out_size; (void)ws_size;
  const float* x      = (const float*)d_in[0];
  const float* w_in   = (const float*)d_in[1];
  const float* bn1_g  = (const float*)d_in[2];
  const float* bn1_b  = (const float*)d_in[3];
  const float* bn1_m  = (const float*)d_in[4];
  const float* bn1_v  = (const float*)d_in[5];
  const float* w_qkv  = (const float*)d_in[6];
  const float* rel_h  = (const float*)d_in[7];
  const float* rel_w  = (const float*)d_in[8];
  const float* w_out  = (const float*)d_in[9];
  const float* bn2_g  = (const float*)d_in[10];
  const float* bn2_b  = (const float*)d_in[11];
  const float* bn2_m  = (const float*)d_in[12];
  const float* bn2_v  = (const float*)d_in[13];
  const float* w_head = (const float*)d_in[14];
  const float* b_head = (const float*)d_in[15];
  float* out = (float*)d_out;

  char* ws = (char*)d_ws;
  const size_t QKV_BYTES  = (size_t)HEADS * HWSZ * DH * 2;   // bf16
  unsigned short* qb = (unsigned short*)(ws);                        // 4.5 MB
  unsigned short* kb = (unsigned short*)(ws + QKV_BYTES);
  unsigned short* vb = (unsigned short*)(ws + 2 * QKV_BYTES);
  float* RW     = (float*)(ws + 3 * QKV_BYTES);                      // 7.1 MB
  float* RH     = (float*)((char*)RW + (size_t)HEADS * HWSZ * WW * 4); // 3.5 MB
  float* h1     = (float*)((char*)RH + (size_t)HEADS * HWSZ * HH * 4); // 1.2 MB
  float* attn_t = (float*)((char*)h1 + (size_t)64 * HWSZ * 4);       // 9.4 MB
  float* fm     = (float*)((char*)attn_t + (size_t)HEADS * DH * HWSZ * 4); // 4.7 MB

  const int GX = HWSZ / 256;   // 18

  k_proj_in <<<dim3(GX, 64),   256, 0, stream>>>(x, w_in, bn1_g, bn1_b, bn1_m, bn1_v, h1);
  k_qkv     <<<dim3(GX, 1536), 256, 0, stream>>>(h1, w_qkv, qb, kb, vb);
  k_relbias <<<dim3(GX, HEADS * 144), 256, 0, stream>>>(qb, rel_h, rel_w, RW, RH);
  k_attn    <<<dim3(HEADS * 72), 128, 0, stream>>>(qb, kb, vb, RW, RH, attn_t);
  k_out_proj<<<dim3(GX, 256),  256, 0, stream>>>(attn_t, w_out, bn2_g, bn2_b, bn2_m, bn2_v, x, fm);
  k_head    <<<dim3(GX, 19),   256, 0, stream>>>(fm, w_head, b_head, out);
}
